// Up_25786983645747
// MI455X (gfx1250) — matrix operand order repaired and checked by reading, not compiled, not run on hardware
//
#include <hip/hip_runtime.h>
#include <hip/hip_bf16.h>
#include <cstdint>

typedef float  v8f   __attribute__((ext_vector_type(8)));
typedef __bf16 v16bf __attribute__((ext_vector_type(16)));
typedef __bf16 bf16x8 __attribute__((ext_vector_type(8)));

__device__ __forceinline__ v16bf cat8(bf16x8 lo, bf16x8 hi) {
    return __builtin_shufflevector(lo, hi, 0,1,2,3,4,5,6,7,8,9,10,11,12,13,14,15);
}

// ---------------------------------------------------------------------------
// g[b,co] = SCALE*style * rsqrt(SCALE^2*style^2*sum(w_style[co]^2) + 1e-8)
// ---------------------------------------------------------------------------
__global__ void prep_g_kernel(const float* __restrict__ xg,
                              const float* __restrict__ w_mod,
                              const float* __restrict__ b_mod,
                              const float* __restrict__ w_style,
                              float* __restrict__ g) {
    int b = blockIdx.x, co = threadIdx.x;            // 8 x 128
    float s = b_mod[co];
    const float* xgb = xg + b * 512;
    const float* wm  = w_mod + co * 512;
    for (int k = 0; k < 512; ++k) s += xgb[k] * wm[k];
    float w2 = 0.f;
    const float* wsr = w_style + co * 1152;
    for (int j = 0; j < 1152; ++j) { float w = wsr[j]; w2 += w * w; }
    const float SCALE = 0.02946278254943948f;        // 1/sqrt(1152)
    float t = SCALE * s;
    g[b * 128 + co] = t * rsqrtf(t * t * w2 + 1e-8f);
}

// ---------------------------------------------------------------------------
// Repack fp32 conv weights [Co][Cin][k][k] into bf16 WMMA A-fragment order
// frag[kpos][chunk][mt][lane][16] per the 16x32 bf16 A layout.
// ---------------------------------------------------------------------------
__global__ void prep_wfrag_kernel(const float* __restrict__ w,
                                  __bf16* __restrict__ frag,
                                  int Cin, int ksz) {
    int t = blockIdx.x * blockDim.x + threadIdx.x;
    int nchunk = Cin >> 5;
    int total = ksz * ksz * nchunk * 256;
    if (t >= total) return;
    int lane  = t & 31;
    int mt    = (t >> 5) & 7;
    int rest  = t >> 8;
    int chunk = rest % nchunk;
    int kpos  = rest / nchunk;
    int kh = kpos / ksz, kw = kpos % ksz;
    int co = mt * 16 + (lane & 15);
    int kadd = (lane < 16) ? 0 : 8;
    __bf16* o = frag + (size_t)t * 16;
    #pragma unroll
    for (int i = 0; i < 16; ++i) {
        int kl = (i < 8) ? (kadd + i) : (16 + kadd + (i - 8));
        int ci = chunk * 32 + kl;
        o[i] = (__bf16)w[(((size_t)co * Cin + ci) * ksz + kh) * ksz + kw];
    }
}

// ---------------------------------------------------------------------------
// Build NHWC bf16 concat input: xin[b][y][x][c], c<128 -> x2, c>=128 -> 2x
// bilinear upsample of x1 (half-pixel, edge clamp). LDS transpose keeps both
// the NCHW reads and NHWC writes coalesced.
// ---------------------------------------------------------------------------
__global__ __launch_bounds__(256) void prep_xin_kernel(
        const float* __restrict__ x1, const float* __restrict__ x2,
        __bf16* __restrict__ xin) {
    __shared__ float tile[32 * 132];
    int b = blockIdx.y, y = blockIdx.x, tid = threadIdx.x;
    // ---- x2 channels ----
    for (int c0 = 0; c0 < 128; c0 += 32) {
        for (int i = tid; i < 4096; i += 256) {
            int ci = i >> 7, x = i & 127;
            tile[ci * 132 + x] = x2[(((size_t)b * 128 + c0 + ci) << 14) + (y << 7) + x];
        }
        __syncthreads();
        for (int i = tid; i < 4096; i += 256) {
            int ci = i & 31, x = i >> 5;
            xin[((size_t)((b * 128 + y) * 128 + x)) * 256 + c0 + ci] = (__bf16)tile[ci * 132 + x];
        }
        __syncthreads();
    }
    // ---- upsampled x1 channels (two source rows per output row) ----
    int ym = y >> 1; int y0, y1v; float wy0, wy1;
    if ((y & 1) == 0) { y0 = ym - 1; y1v = ym;     wy0 = 0.25f; wy1 = 0.75f; }
    else              { y0 = ym;     y1v = ym + 1; wy0 = 0.75f; wy1 = 0.25f; }
    y0 = max(y0, 0); y1v = min(y1v, 63);
    for (int c0 = 0; c0 < 128; c0 += 32) {
        for (int i = tid; i < 4096; i += 256) {
            int ci = i >> 7, j = i & 127, row = j >> 6, xs = j & 63;
            int ys = row ? y1v : y0;
            tile[ci * 132 + row * 64 + xs] =
                x1[(((size_t)b * 128 + c0 + ci) << 12) + (ys << 6) + xs];
        }
        __syncthreads();
        for (int i = tid; i < 4096; i += 256) {
            int ci = i & 31, x = i >> 5;
            int xm = x >> 1; int x0, x1i; float wx0, wx1;
            if ((x & 1) == 0) { x0 = xm - 1; x1i = xm;     wx0 = 0.25f; wx1 = 0.75f; }
            else              { x0 = xm;     x1i = xm + 1; wx0 = 0.75f; wx1 = 0.25f; }
            x0 = max(x0, 0); x1i = min(x1i, 63);
            const float* r0 = &tile[ci * 132];
            float v = wy0 * (wx0 * r0[x0] + wx1 * r0[x1i]) +
                      wy1 * (wx0 * r0[64 + x0] + wx1 * r0[64 + x1i]);
            xin[((size_t)((b * 128 + y) * 128 + x)) * 256 + 128 + c0 + ci] = (__bf16)v;
        }
        __syncthreads();
    }
}

// ---------------------------------------------------------------------------
// Implicit-GEMM conv, NHWC activations, double-buffered LDS staged with
// GLOBAL_LOAD_ASYNC_TO_LDS_B128 (ASYNCcnt), compute via
// v_wmma_f32_16x16x32_bf16. Block = (b, row y), 8 waves; wave = co tile.
// ---------------------------------------------------------------------------
template <int KSIZE, int NCHUNK>
__global__ __launch_bounds__(256) void conv_wmma_kernel(
    const __bf16* __restrict__ in,     // [B][128][128][CIN] bf16 NHWC
    const __bf16* __restrict__ wfrag,  // pre-fragmented bf16 weights
    const float*  __restrict__ bias,   // [128] or null
    const float*  __restrict__ gmod,   // [B][128] or null
    const float*  __restrict__ resid,  // [B][128][128][128] f32 NHWC or null
    float*        __restrict__ outf,   // f32 out or null
    __bf16*       __restrict__ outh,   // bf16 NHWC out or null
    int leaky, float pscale, int out_nchw)
{
    constexpr int HALO = (KSIZE == 3) ? 1 : 0;
    constexpr int ROWS = 2 * HALO + 1;
    constexpr int COLS = 128 + 2 * HALO;
    constexpr int CSTR = 48;               // 32 ci + pad -> 96B/col, 16B aligned
    constexpr int CIN  = NCHUNK * 32;
    constexpr int TILE = ROWS * COLS * CSTR;
    __shared__ __align__(16) __bf16 lds[2 * TILE];

    const int b = blockIdx.y, y = blockIdx.x;
    const int tid = threadIdx.x, wave = tid >> 5, lane = tid & 31;
    const int kbase = (lane < 16) ? 0 : 8;
    const int n = lane & 15;

    const unsigned long long gbase =
        (unsigned long long)(uintptr_t)(in + (size_t)b * 128 * 128 * CIN);

    // Static zero fill: x-halo columns (always outside image) and
    // out-of-image halo rows, in BOTH ping-pong buffers (never restaged).
    if (KSIZE == 3) {
        for (int i = tid; i < 2 * ROWS * 2 * 32; i += 256) {   // 384
            int ci = i & 31; int rest = i >> 5;
            int side = rest & 1; int rest2 = rest >> 1;
            int r = rest2 % ROWS; int bf = rest2 / ROWS;
            lds[bf * TILE + (r * COLS + side * (COLS - 1)) * CSTR + ci] = (__bf16)0.f;
        }
        if (y == 0)
            for (int i = tid; i < 2 * 128 * 32; i += 256) {
                int ci = i & 31, col = (i >> 5) & 127, bf = i >> 12;
                lds[bf * TILE + (0 * COLS + col + 1) * CSTR + ci] = (__bf16)0.f;
            }
        if (y == 127)
            for (int i = tid; i < 2 * 128 * 32; i += 256) {
                int ci = i & 31, col = (i >> 5) & 127, bf = i >> 12;
                lds[bf * TILE + (2 * COLS + col + 1) * CSTR + ci] = (__bf16)0.f;
            }
    }

    // Async-stage one 32-channel chunk: 64B contiguous per (row,col), 4 x b128.
    auto stage = [&](int chunk, int bufsel) {
        const int NT = ROWS * 128 * 4;
        for (int t = tid; t < NT; t += 256) {          // <= 6 asyncs per wave
            int q = t & 3, col = (t >> 2) & 127, r = t >> 9;
            int gy = y + r - HALO;
            if ((unsigned)gy < 128u) {
                unsigned loff = (unsigned)(uintptr_t)
                    &lds[bufsel * TILE + (r * COLS + col + HALO) * CSTR + q * 8];
                unsigned voff = (unsigned)(((gy * 128 + col) * CIN + chunk * 32) * 2 + q * 16);
                asm volatile("global_load_async_to_lds_b128 %0, %1, %2"
                             :: "v"(loff), "v"(voff), "s"(gbase) : "memory");
            }
        }
    };

    v8f acc[8] = {};

    stage(0, 0);
    asm volatile("s_wait_asynccnt 0" ::: "memory");
    __syncthreads();

    for (int chunk = 0; chunk < NCHUNK; ++chunk) {
        if (chunk + 1 < NCHUNK) stage(chunk + 1, (chunk + 1) & 1);  // overlap
        const __bf16* buf = &lds[(chunk & 1) * TILE];
        #pragma unroll
        for (int kh = 0; kh < KSIZE; ++kh)
        #pragma unroll
        for (int kw = 0; kw < KSIZE; ++kw) {
            int kpos = kh * KSIZE + kw;
            const __bf16* wp =
                wfrag + ((((size_t)kpos * NCHUNK + chunk) * 8 + wave) * 32 + lane) * 16;
            v16bf a = cat8(*(const bf16x8*)wp, *(const bf16x8*)(wp + 8));
            #pragma unroll
            for (int nt = 0; nt < 8; ++nt) {
                int px = nt * 16 + n + kw;
                const __bf16* bp = &buf[(kh * COLS + px) * CSTR + kbase];
                v16bf bb = cat8(*(const bf16x8*)bp, *(const bf16x8*)(bp + 16));
                acc[nt] = __builtin_amdgcn_wmma_f32_16x16x32_bf16(
                    false, a, false, bb, (short)0, acc[nt], false, false);
            }
        }
        asm volatile("s_wait_asynccnt 0" ::: "memory");
        __syncthreads();
    }

    // Epilogue. D layout: VGPR r -> M=r (lanes 0-15) / M=8+r (lanes 16-31).
    int rofs = (lane < 16) ? 0 : 8;
    #pragma unroll
    for (int r = 0; r < 8; ++r) {
        int co = wave * 16 + rofs + r;
        float bi = bias ? bias[co] : 0.f;
        float gg = gmod ? gmod[b * 128 + co] : 1.f;
        #pragma unroll
        for (int nt = 0; nt < 8; ++nt) {
            int px = nt * 16 + n;
            size_t nhwc = ((size_t)((b * 128 + y) * 128 + px)) * 128 + co;
            float v = acc[nt][r] + bi;
            if (resid) v += resid[nhwc];
            v *= pscale;
            v *= gg;
            if (leaky) v = (v >= 0.f) ? v : 0.1f * v;
            if (outh) outh[nhwc] = (__bf16)v;
            if (outf) {
                size_t oi = out_nchw
                    ? ((((size_t)b * 128 + co) << 14) + (y << 7) + px)
                    : nhwc;
                outf[oi] = v;
            }
        }
    }
}

// ---------------------------------------------------------------------------
extern "C" void kernel_launch(void* const* d_in, const int* in_sizes, int n_in,
                              void* d_out, int out_size, void* d_ws, size_t ws_size,
                              hipStream_t stream) {
    (void)in_sizes; (void)n_in; (void)out_size; (void)ws_size;
    const float* x1       = (const float*)d_in[0];
    const float* x2       = (const float*)d_in[1];
    const float* xg       = (const float*)d_in[2];
    const float* w_mod    = (const float*)d_in[3];
    const float* b_mod    = (const float*)d_in[4];
    const float* w_style  = (const float*)d_in[5];
    const float* w_bottle = (const float*)d_in[6];
    const float* b_bottle = (const float*)d_in[7];
    const float* w_dc1    = (const float*)d_in[8];
    const float* b_dc1    = (const float*)d_in[9];
    const float* w_dc2    = (const float*)d_in[10];
    const float* b_dc2    = (const float*)d_in[11];
    const float* w_last   = (const float*)d_in[12];
    const float* b_last   = (const float*)d_in[13];

    char* ws = (char*)d_ws;
    size_t cur = 0;
    auto take = [&](size_t bytes) -> char* {
        char* p = ws + cur;
        cur += (bytes + 255) & ~(size_t)255;
        return p;
    };
    const size_t ACT  = (size_t)8 * 128 * 128 * 128;
    float*  gbuf = (float*)take(8 * 128 * sizeof(float));
    __bf16* wfB  = (__bf16*)take((size_t)1 * 8 * 256 * 16 * 2);
    __bf16* wf1  = (__bf16*)take((size_t)9 * 4 * 256 * 16 * 2);
    __bf16* wf2  = (__bf16*)take((size_t)9 * 4 * 256 * 16 * 2);
    __bf16* wfS  = (__bf16*)take((size_t)9 * 4 * 256 * 16 * 2);
    __bf16* wfL  = (__bf16*)take((size_t)9 * 4 * 256 * 16 * 2);
    __bf16* xin  = (__bf16*)take((size_t)8 * 256 * 128 * 128 * 2); // NHWC C=256
    float*  xres = (float*)take(ACT * 4);                          // NHWC f32
    __bf16* bufA = (__bf16*)take(ACT * 2);                         // NHWC bf16
    __bf16* bufB = (__bf16*)take(ACT * 2);

    prep_g_kernel<<<8, 128, 0, stream>>>(xg, w_mod, b_mod, w_style, gbuf);
    prep_wfrag_kernel<<<8, 256, 0, stream>>>(w_bottle, wfB, 256, 1);
    prep_wfrag_kernel<<<36, 256, 0, stream>>>(w_dc1, wf1, 128, 3);
    prep_wfrag_kernel<<<36, 256, 0, stream>>>(w_dc2, wf2, 128, 3);
    prep_wfrag_kernel<<<36, 256, 0, stream>>>(w_style, wfS, 128, 3);
    prep_wfrag_kernel<<<36, 256, 0, stream>>>(w_last, wfL, 128, 3);
    prep_xin_kernel<<<dim3(128, 8), 256, 0, stream>>>(x1, x2, xin);

    dim3 cg(128, 8);  // (row, batch)
    // bottleneck 1x1 (256->128): f32 residual base + bf16 for dc1
    conv_wmma_kernel<1, 8><<<cg, 256, 0, stream>>>(
        xin, wfB, b_bottle, nullptr, nullptr, xres, bufA, 0, 1.f, 0);
    // dc1 3x3 + leaky
    conv_wmma_kernel<3, 4><<<cg, 256, 0, stream>>>(
        bufA, wf1, b_dc1, nullptr, nullptr, nullptr, bufB, 1, 1.f, 0);
    // dc2 3x3 + residual, * 1/sqrt(2)
    conv_wmma_kernel<3, 4><<<cg, 256, 0, stream>>>(
        bufB, wf2, b_dc2, nullptr, xres, nullptr, bufA, 0, 0.70710678118f, 0);
    // modulated conv == shared w_style conv scaled by g[b,co]
    conv_wmma_kernel<3, 4><<<cg, 256, 0, stream>>>(
        bufA, wfS, nullptr, gbuf, nullptr, nullptr, bufB, 0, 1.f, 0);
    // conv_last 3x3 + leaky -> NCHW f32 d_out
    conv_wmma_kernel<3, 4><<<cg, 256, 0, stream>>>(
        bufB, wfL, b_last, nullptr, nullptr, (float*)d_out, nullptr, 1, 1.f, 1);
}
